// MultiheadSelfAttention_3917010173994
// MI455X (gfx1250) — matrix operand order repaired and checked by reading, not compiled, not run on hardware
//
#include <hip/hip_runtime.h>

#define B_ 4
#define S_ 2048
#define D_ 1024
#define H_ 16
#define HD_ 64

typedef __attribute__((ext_vector_type(16))) __bf16 v16bf;
typedef __attribute__((ext_vector_type(8)))  float  v8f;
typedef __attribute__((ext_vector_type(4)))  int    v4i;

#if defined(__has_builtin)
#if __has_builtin(__builtin_amdgcn_global_load_async_to_lds_b128) && \
    __has_builtin(__builtin_amdgcn_s_wait_asynccnt)
#define ASYNC_CP 1
#endif
#endif

// 16-byte global -> LDS copy. gfx1250 async-to-LDS (ASYNCcnt) when available.
__device__ __forceinline__ void cp_b128(__bf16* lds, const __bf16* g) {
#if defined(ASYNC_CP)
  __builtin_amdgcn_global_load_async_to_lds_b128((v4i*)g, (v4i*)lds, 0, 0);
#else
  *(int4*)lds = *(const int4*)g;
#endif
}

// Wait until at most N async copies remain outstanding (in-order completion:
// older tiles drain first, newer prefetches stay in flight).
template <int N>
__device__ __forceinline__ void async_wait_le() {
#if defined(ASYNC_CP)
  __builtin_amdgcn_s_wait_asynccnt(N);
#endif
}

__device__ __forceinline__ v8f wmma_bf16(v16bf a, v16bf b, v8f c) {
  return __builtin_amdgcn_wmma_f32_16x16x32_bf16(false, a, false, b, (short)0, c, false, false);
}

// A-matrix fragment (16x32 bf16) from LDS tile stored [row][k].
// K order used here (matches B loaders): lanes 0-15 -> M=lane, K 0-15; lanes 16-31 -> K 16-31 (contiguous).
__device__ __forceinline__ v16bf load_a_frag_lds(const __bf16* src, int stride) {
  const int lane = threadIdx.x & 31;
  const int row  = lane & 15;
  const int kg   = (lane >> 4) << 3;
  v16bf a;
#pragma unroll
  for (int e = 0; e < 16; ++e) {
    const int j = e >> 1, w = e & 1;
    const int k = (kg << 1) + e;
    a[e] = src[row * stride + k];
  }
  return a;
}

// Same A layout from global bf16 with (power-of-two) scale folded in.
__device__ __forceinline__ v16bf load_a_frag_global(const __bf16* src, int stride,
                                                    int koff, float scale) {
  const int lane = threadIdx.x & 31;
  const int row  = lane & 15;
  const int kg   = (lane >> 4) << 3;
  v16bf a;
#pragma unroll
  for (int e = 0; e < 16; ++e) {
    const int j = e >> 1, w = e & 1;
    const int k = (kg << 1) + e;
    a[e] = (__bf16)((float)src[(long)row * stride + koff + k] * scale);
  }
  return a;
}

// B fragment (32x16), LDS holds [n][k]: B[k][n] = lds[n*stride + k].
__device__ __forceinline__ v16bf load_b_frag_nk(const __bf16* src, int stride,
                                                int noff, int koff) {
  const int lane = threadIdx.x & 31;
  const int n  = noff + (lane & 15);
  const int kb = koff + ((lane >> 4) << 4);
  v16bf b;
#pragma unroll
  for (int e = 0; e < 16; ++e) b[e] = src[n * stride + kb + e];
  return b;
}

// B fragment (32x16), LDS holds [k][n].
__device__ __forceinline__ v16bf load_b_frag_kn(const __bf16* src, int stride,
                                                int koff, int noff) {
  const int lane = threadIdx.x & 31;
  const int n  = noff + (lane & 15);
  const int kb = koff + ((lane >> 4) << 4);
  v16bf b;
#pragma unroll
  for (int e = 0; e < 16; ++e) b[e] = src[(kb + e) * stride + n];
  return b;
}

// ---------------------------------------------------------------------------
// One-time fp32 -> bf16 conversion (x and weights), 4 elems/thread.
// ---------------------------------------------------------------------------
__global__ __launch_bounds__(256) void cvt_f32_bf16_kernel(
    const float* __restrict__ src, __bf16* __restrict__ dst, long n) {
  const long i = ((long)blockIdx.x * 256 + threadIdx.x) * 4;
  if (i + 3 >= n) {
    for (long j = i; j < n; ++j) dst[j] = (__bf16)src[j];
    return;
  }
  const float4 v = *(const float4*)(src + i);
  dst[i + 0] = (__bf16)v.x;
  dst[i + 1] = (__bf16)v.y;
  dst[i + 2] = (__bf16)v.z;
  dst[i + 3] = (__bf16)v.w;
}

// ---------------------------------------------------------------------------
// C[M x N] = A[M x K] * W[N x K]^T   (bf16 in, bf16 or fp32 out)
// 256 threads (8 waves), 128x64 tile, 32x32 per wave (4 WMMAs/step).
// Double-buffered async staging: tile i+1 copies overlap tile i compute.
// ---------------------------------------------------------------------------
template <bool OUT_BF16>
__global__ __launch_bounds__(256) void gemm_bf16_kernel(
    const __bf16* __restrict__ A, const __bf16* __restrict__ W,
    void* __restrict__ Cout, int M, int N, int K) {
  __shared__ __bf16 sA[2][128 * 32];  // 2 x 8 KB
  __shared__ __bf16 sW[2][64 * 32];   // 2 x 4 KB
  const int tiles_n = N >> 6;
  const int bm = blockIdx.x / tiles_n;
  const int bn = blockIdx.x % tiles_n;
  const int tid = threadIdx.x;
  const int wave = tid >> 5;
  const int wm = (wave >> 1) << 5;  // 0,32,64,96
  const int wn = (wave & 1) << 5;   // 0,32
  const long rowA = (long)bm * 128;
  const long colC = (long)bn * 64;
  v8f acc00 = {}, acc01 = {}, acc10 = {}, acc11 = {};
  const int rA = tid >> 1, cA = (tid & 1) * 16;   // A tile: 128 x 32
  const int rW = tid >> 2, cW = (tid & 3) * 8;    // W tile:  64 x 32
  // 3 async b128 copies per thread per tile
  auto stage = [&](int k0, int buf) {
    const __bf16* gA = A + (rowA + rA) * (long)K + k0 + cA;
    cp_b128(&sA[buf][rA * 32 + cA], gA);
    cp_b128(&sA[buf][rA * 32 + cA + 8], gA + 8);
    cp_b128(&sW[buf][rW * 32 + cW], W + (colC + rW) * (long)K + k0 + cW);
  };
  stage(0, 0);
  for (int k0 = 0; k0 < K; k0 += 32) {
    const int buf = (k0 >> 5) & 1;
    if (k0 + 32 < K) {
      stage(k0 + 32, buf ^ 1);   // prefetch next tile into other buffer
      async_wait_le<3>();        // drain this tile's 3 copies; keep next 3 in flight
    } else {
      async_wait_le<0>();
    }
    __syncthreads();
    const v16bf a0 = load_a_frag_lds(&sA[buf][wm * 32], 32);
    const v16bf a1 = load_a_frag_lds(&sA[buf][(wm + 16) * 32], 32);
    const v16bf b0 = load_b_frag_nk(sW[buf], 32, wn, 0);
    const v16bf b1 = load_b_frag_nk(sW[buf], 32, wn + 16, 0);
    acc00 = wmma_bf16(a0, b0, acc00);
    acc01 = wmma_bf16(a0, b1, acc01);
    acc10 = wmma_bf16(a1, b0, acc10);
    acc11 = wmma_bf16(a1, b1, acc11);
    __syncthreads();
  }
  const int lane = tid & 31;
  const int lcol = lane & 15, lhi = lane >> 4;
#pragma unroll
  for (int r = 0; r < 8; ++r) {
    const long row0 = rowA + wm + r + 8 * lhi;
    const long row1 = row0 + 16;
    const long c0 = colC + wn + lcol, c1 = c0 + 16;
    if (OUT_BF16) {
      __bf16* C = (__bf16*)Cout;
      C[row0 * N + c0] = (__bf16)acc00[r];
      C[row0 * N + c1] = (__bf16)acc01[r];
      C[row1 * N + c0] = (__bf16)acc10[r];
      C[row1 * N + c1] = (__bf16)acc11[r];
    } else {
      float* C = (float*)Cout;
      C[row0 * N + c0] = acc00[r];
      C[row0 * N + c1] = acc01[r];
      C[row1 * N + c0] = acc10[r];
      C[row1 * N + c1] = acc11[r];
    }
  }
}

// ---------------------------------------------------------------------------
// RoPE in place on bf16 Q and K ([B*S, D] layout).
// ---------------------------------------------------------------------------
__global__ __launch_bounds__(256) void rope_kernel(__bf16* __restrict__ Q,
                                                   __bf16* __restrict__ K) {
  const long idx = (long)blockIdx.x * 256 + threadIdx.x;  // one per pair
  const int PH = D_ / 2;
  const int row = (int)(idx / PH);
  const int p   = (int)(idx % PH);
  const int s   = row & (S_ - 1);
  const int i   = p & (HD_ / 2 - 1);
  const int h   = p >> 5;
  const float inv = __expf(-(float)i * (9.210340371976184f / 32.0f));  // theta^-i/32
  float sn, cs;
  __sincosf((float)s * inv, &sn, &cs);
  const long off = (long)row * D_ + (long)h * HD_ + 2 * i;
  const float q1 = (float)Q[off], q2 = (float)Q[off + 1];
  Q[off]     = (__bf16)(q1 * cs - q2 * sn);
  Q[off + 1] = (__bf16)(q1 * sn + q2 * cs);
  const float k1 = (float)K[off], k2 = (float)K[off + 1];
  K[off]     = (__bf16)(k1 * cs - k2 * sn);
  K[off + 1] = (__bf16)(k1 * sn + k2 * cs);
}

// ---------------------------------------------------------------------------
// Flash attention: grid (S/128, H, B). 8 waves, wave w owns 16 q rows.
// Double-buffered async K/V staging (32-key chunks); online softmax with
// wave32 shfl row reductions; P bounced through per-wave LDS to A-layout.
// All four B fragments are loaded into distinct registers BEFORE the WMMAs
// so the ds_loads clause together instead of serializing on s_wait_dscnt 0.
// ---------------------------------------------------------------------------
__global__ __launch_bounds__(256) void attn_kernel(
    const __bf16* __restrict__ Q, const __bf16* __restrict__ Kt,
    const __bf16* __restrict__ Vt, __bf16* __restrict__ O) {
  __shared__ __bf16 sK[2][32 * 64];
  __shared__ __bf16 sV[2][32 * 64];
  __shared__ __bf16 sP[8 * 16 * 32];
  const int qblk = blockIdx.x;
  const int h    = blockIdx.y;
  const int b    = blockIdx.z;
  const int tid  = threadIdx.x;
  const int wave = tid >> 5;
  const int lane = tid & 31;
  const int lcol = lane & 15, lhi = lane >> 4;
  const int q0   = qblk * 128 + wave * 16;
  const long base = (long)b * S_ * D_;
  const int ch    = h * HD_;
  const __bf16* qp = Q + base + (long)q0 * D_ + ch;
  const v16bf qf0 = load_a_frag_global(qp, D_, 0, 0.125f);   // 1/sqrt(64) exact
  const v16bf qf1 = load_a_frag_global(qp, D_, 32, 0.125f);
  v8f o0 = {}, o1 = {}, o2 = {}, o3 = {};
  float mrow[8], lrow[8];
#pragma unroll
  for (int r = 0; r < 8; ++r) { mrow[r] = -1e30f; lrow[r] = 0.0f; }
  __bf16* myP = sP + wave * (16 * 32);
  const int rT = tid >> 3, cT = (tid & 7) * 8;  // staging coords: 32 x 64 tile
  auto stage = [&](int k0, int buf) {
    cp_b128(&sK[buf][rT * 64 + cT], Kt + base + (long)(k0 + rT) * D_ + ch + cT);
    cp_b128(&sV[buf][rT * 64 + cT], Vt + base + (long)(k0 + rT) * D_ + ch + cT);
  };
  const int kend = qblk * 128 + 128;
  stage(0, 0);
  for (int k0 = 0; k0 < kend; k0 += 32) {
    const int buf = (k0 >> 5) & 1;
    if (k0 + 32 < kend) {
      stage(k0 + 32, buf ^ 1);   // prefetch next K/V chunk
      async_wait_le<2>();        // drain this chunk's 2 copies
    } else {
      async_wait_le<0>();
    }
    __syncthreads();
    // scores (16q x 32keys): B[k=hd][n=key] = sK[key][hd].
    // Load all four fragments first, then issue the WMMAs.
    const v16bf bk00 = load_b_frag_nk(sK[buf], 64, 0, 0);
    const v16bf bk01 = load_b_frag_nk(sK[buf], 64, 0, 32);
    const v16bf bk10 = load_b_frag_nk(sK[buf], 64, 16, 0);
    const v16bf bk11 = load_b_frag_nk(sK[buf], 64, 16, 32);
    v8f s0 = {}, s1 = {};
    s0 = wmma_bf16(qf0, bk00, s0);
    s0 = wmma_bf16(qf1, bk01, s0);
    s1 = wmma_bf16(qf0, bk10, s1);
    s1 = wmma_bf16(qf1, bk11, s1);
#pragma unroll
    for (int r = 0; r < 8; ++r) {
      const int qrow = q0 + r + 8 * lhi;
      const int key0 = k0 + lcol, key1 = k0 + 16 + lcol;
      const float v0 = (key0 <= qrow) ? s0[r] : -1e30f;
      const float v1 = (key1 <= qrow) ? s1[r] : -1e30f;
      float mx = fmaxf(v0, v1);
#pragma unroll
      for (int off = 1; off < 16; off <<= 1)
        mx = fmaxf(mx, __shfl_xor(mx, off, 32));
      const float mn   = fmaxf(mrow[r], mx);
      const float corr = __expf(mrow[r] - mn);
      const float p0 = __expf(v0 - mn);
      const float p1 = __expf(v1 - mn);
      float ps = p0 + p1;
#pragma unroll
      for (int off = 1; off < 16; off <<= 1)
        ps += __shfl_xor(ps, off, 32);
      lrow[r] = lrow[r] * corr + ps;
      mrow[r] = mn;
      o0[r] *= corr; o1[r] *= corr; o2[r] *= corr; o3[r] *= corr;
      myP[(r + 8 * lhi) * 32 + lcol]      = (__bf16)p0;
      myP[(r + 8 * lhi) * 32 + 16 + lcol] = (__bf16)p1;
    }
    // O += P(16x32) @ V(32 x 64): B[k=key][n=hd] = sV[key][hd].
    const v16bf pf  = load_a_frag_lds(myP, 32);
    const v16bf bv0 = load_b_frag_kn(sV[buf], 64, 0, 0);
    const v16bf bv1 = load_b_frag_kn(sV[buf], 64, 0, 16);
    const v16bf bv2 = load_b_frag_kn(sV[buf], 64, 0, 32);
    const v16bf bv3 = load_b_frag_kn(sV[buf], 64, 0, 48);
    o0 = wmma_bf16(pf, bv0, o0);
    o1 = wmma_bf16(pf, bv1, o1);
    o2 = wmma_bf16(pf, bv2, o2);
    o3 = wmma_bf16(pf, bv3, o3);
    __syncthreads();
  }
#pragma unroll
  for (int r = 0; r < 8; ++r) {
    const float inv = 1.0f / lrow[r];
    const long ro = base + (long)(q0 + r + 8 * lhi) * D_ + ch;
    O[ro + lcol]      = (__bf16)(o0[r] * inv);
    O[ro + 16 + lcol] = (__bf16)(o1[r] * inv);
    O[ro + 32 + lcol] = (__bf16)(o2[r] * inv);
    O[ro + 48 + lcol] = (__bf16)(o3[r] * inv);
  }
}

extern "C" void kernel_launch(void* const* d_in, const int* in_sizes, int n_in,
                              void* d_out, int out_size, void* d_ws, size_t ws_size,
                              hipStream_t stream) {
  const float* x  = (const float*)d_in[0];
  const float* WQ = (const float*)d_in[1];
  const float* WK = (const float*)d_in[2];
  const float* WV = (const float*)d_in[3];
  const float* WO = (const float*)d_in[4];
  float* out = (float*)d_out;
  const long NX = (long)B_ * S_ * D_;  // 8M
  const long NW = (long)D_ * D_;       // 1M
  __bf16* xb = (__bf16*)d_ws;
  __bf16* wq = xb + NX;
  __bf16* wk = wq + NW;
  __bf16* wv = wk + NW;
  __bf16* wo = wv + NW;
  __bf16* Q  = wo + NW;
  __bf16* K  = Q + NX;
  __bf16* V  = K + NX;
  __bf16* O  = V + NX;
  dim3 blk(256);
  cvt_f32_bf16_kernel<<<dim3((unsigned)(NX / 4 / 256)), blk, 0, stream>>>(x, xb, NX);
  cvt_f32_bf16_kernel<<<dim3((unsigned)(NW / 4 / 256)), blk, 0, stream>>>(WQ, wq, NW);
  cvt_f32_bf16_kernel<<<dim3((unsigned)(NW / 4 / 256)), blk, 0, stream>>>(WK, wk, NW);
  cvt_f32_bf16_kernel<<<dim3((unsigned)(NW / 4 / 256)), blk, 0, stream>>>(WV, wv, NW);
  cvt_f32_bf16_kernel<<<dim3((unsigned)(NW / 4 / 256)), blk, 0, stream>>>(WO, wo, NW);
  const int M = B_ * S_;
  dim3 gg((M / 128) * (D_ / 64));  // 1024 blocks
  gemm_bf16_kernel<true><<<gg, blk, 0, stream>>>(xb, wq, Q, M, D_, D_);
  gemm_bf16_kernel<true><<<gg, blk, 0, stream>>>(xb, wk, K, M, D_, D_);
  gemm_bf16_kernel<true><<<gg, blk, 0, stream>>>(xb, wv, V, M, D_, D_);
  rope_kernel<<<dim3((unsigned)(NX / 2 / 256)), blk, 0, stream>>>(Q, K);
  attn_kernel<<<dim3(S_ / 128, H_, B_), blk, 0, stream>>>(Q, K, V, O);
  gemm_bf16_kernel<false><<<gg, blk, 0, stream>>>(O, wo, out, M, D_, D_);
}
